// WSimNet_57080115364196
// MI455X (gfx1250) — matrix operand order repaired and checked by reading, not compiled, not run on hardware
//
#include <hip/hip_runtime.h>
#include <cstdint>

// ---------------------------------------------------------------- types
typedef __attribute__((ext_vector_type(16))) __bf16 v16bf;
typedef __attribute__((ext_vector_type(8)))  float  v8f;

union Frag {            // 16 bf16 = two 16-byte chunks
    v16bf v;
    uint4 q[2];
};

static __device__ __forceinline__ int lbound(const int* a, int n, int v) {
    int lo = 0, hi = n;
    while (lo < hi) { int mid = (lo + hi) >> 1; if (a[mid] < v) lo = mid + 1; else hi = mid; }
    return lo;
}

// ------------------------------------------------ prep: fp32 -> padded bf16
__global__ __launch_bounds__(256) void pack_x_kernel(const float* __restrict__ x,
                                                     __bf16* __restrict__ out,
                                                     long long total, int C, int CP) {
    long long i = (long long)blockIdx.x * blockDim.x + threadIdx.x;
    if (i >= total) return;
    int c = (int)(i % CP);
    long long m = i / CP;
    out[i] = (__bf16)((c < C) ? x[m * C + c] : 0.0f);
}

// W [K, Nout] fp32  ->  Bt [NP, KP] bf16 (transposed, zero padded)
__global__ __launch_bounds__(256) void pack_wt_kernel(const float* __restrict__ W,
                                                      __bf16* __restrict__ Bt,
                                                      int K, int Nout, int KP, int NP) {
    int idx = blockIdx.x * blockDim.x + threadIdx.x;
    if (idx >= NP * KP) return;
    int n = idx / KP, k = idx % KP;
    float v = (k < K && n < Nout) ? W[(size_t)k * Nout + n] : 0.0f;
    Bt[idx] = (__bf16)v;
}

// fold bias + BN(eval): y = alpha*(x@W) + beta
__global__ __launch_bounds__(256) void make_coef_kernel(const float* __restrict__ b,
                                                        const float* __restrict__ ga,
                                                        const float* __restrict__ be,
                                                        const float* __restrict__ mn,
                                                        const float* __restrict__ vr,
                                                        float* __restrict__ alpha,
                                                        float* __restrict__ betac,
                                                        int C, int CP) {
    int c = blockIdx.x * blockDim.x + threadIdx.x;
    if (c >= CP) return;
    if (c < C) {
        float a = ga[c] * rsqrtf(vr[c] + 1e-5f);
        alpha[c] = a;
        betac[c] = a * (b[c] - mn[c]) + be[c];
    } else {
        alpha[c] = 0.0f;
        betac[c] = 0.0f;
    }
}

// ------------------------------------------------ core WMMA GEMM + BN(+ReLU)
// A  : [M, Kpad] bf16 row-major            (M multiple of 128)
// Bt : [Npad, Kpad] bf16 (pre-transposed)  (Kpad multiple of 64, Npad mult of 128)
// Block: 8 waves, rows [bx*128, +128), cols [by*128, +128).
// B panel staged in LDS via async global->LDS, double-buffered 64-K chunks.
__global__ __launch_bounds__(256)
void gemm_bn_act(const __bf16* __restrict__ A, const __bf16* __restrict__ Bt,
                 const float* __restrict__ alpha, const float* __restrict__ beta,
                 __bf16* __restrict__ outB, float* __restrict__ outF,
                 int M, int Kpad, int ldOut, int relu) {
    __shared__ __align__(16) __bf16 ldsB[2][128 * 64];   // 2 x 16 KB

    const int tid    = threadIdx.x;
    const int lane   = tid & 31;
    const int wave   = tid >> 5;
    const int row0   = (blockIdx.x * 8 + wave) * 16;
    const int colBase = blockIdx.y * 128;

    const int laneHi = lane >> 4;                // 0 | 1
    const int lm     = lane & 15;

    // A fragment (16-bit A 16x32 layout): lanes 0-15 K=0..7,16..23 ; lanes 16-31 K=8..15,24..31
    const __bf16* aRow = A + (size_t)(row0 + lm) * Kpad + laneHi * 8;

    // async fill of one 128x64 B chunk: 1024 16-byte units, 4 per thread
    auto fill = [&](int buf, int k0) {
#pragma unroll
        for (int t = 0; t < 4; ++t) {
            const int e   = tid + t * 256;       // 0..1023
            const int row = e >> 3;              // 0..127 (panel column)
            const int u   = e & 7;               // 16B unit within 64-K row chunk
            const __bf16* src = Bt + (size_t)(colBase + row) * Kpad + k0 + u * 8;
            const uint32_t dst = (uint32_t)(uintptr_t)&ldsB[buf][row * 64 + u * 8];
            asm volatile("global_load_async_to_lds_b128 %0, %1, off"
                         :: "v"(dst), "v"(src) : "memory");
        }
    };

    v8f acc[8] = {};

    fill(0, 0);
    asm volatile("s_wait_asynccnt 0x0" ::: "memory");
    __syncthreads();

    for (int kc = 0; kc < Kpad; kc += 64) {
        const int cur = (kc >> 6) & 1;
        if (kc + 64 < Kpad) {
            fill(cur ^ 1, kc + 64);              // overlap next chunk with compute
            __builtin_prefetch(aRow + kc + 64, 0, 1);   // global_prefetch_b8 for A stream
        }
#pragma unroll
        for (int half = 0; half < 2; ++half) {
            const int k = kc + half * 32;
            Frag a;
            a.q[0] = *(const uint4*)(aRow + k);
            a.q[1] = *(const uint4*)(aRow + k + 16);
#pragma unroll
            for (int t = 0; t < 8; ++t) {
                // B fragment (B 32x16): lane = column n, lanes 0-15 K=0..7,16..23 ; lanes 16-31 K=8..15,24..31
                const __bf16* bp = &ldsB[cur][(t * 16 + lm) * 64 + half * 32 + laneHi * 8];
                Frag b;
                b.q[0] = *(const uint4*)(bp);
                b.q[1] = *(const uint4*)(bp + 16);
                acc[t] = __builtin_amdgcn_wmma_f32_16x16x32_bf16(
                    false, a.v, false, b.v, (short)0, acc[t], false, false);
            }
        }
        asm volatile("s_wait_asynccnt 0x0" ::: "memory");
        __syncthreads();
    }

    if (row0 >= M) return;   // never taken for our launches (M % 128 == 0)

    // D layout: VGPR r, lanes 0-15 -> row r, lanes 16-31 -> row r+8; col = lane&15
    const int rowOut = row0 + laneHi * 8;
#pragma unroll
    for (int t = 0; t < 8; ++t) {
        const int c  = colBase + t * 16 + lm;
        const float al = alpha[c];
        const float be = beta[c];
#pragma unroll
        for (int r = 0; r < 8; ++r) {
            float v = acc[t][r] * al + be;
            if (relu) v = fmaxf(v, 0.0f);
            const size_t off = (size_t)(rowOut + r) * ldOut + c;
            if (outF) outF[off] = v;
            else      outB[off] = (__bf16)v;
        }
    }
}

// ------------------------------------------------ gate scores: per-node dot
__global__ __launch_bounds__(256)
void gate_scores_kernel(const __bf16* __restrict__ g, const float* __restrict__ W2,
                        const float* __restrict__ b2, float* __restrict__ scores,
                        int M, int Kpad, int C) {
    int node = (int)((blockIdx.x * (long long)blockDim.x + threadIdx.x) >> 5);
    int lane = threadIdx.x & 31;
    if (node >= M) return;
    const __bf16* row = g + (size_t)node * Kpad;
    float s = 0.0f;
    for (int c = lane; c < C; c += 32) s += (float)row[c] * W2[c];
#pragma unroll
    for (int off = 16; off; off >>= 1) s += __shfl_xor(s, off, 32);
    if (lane == 0) scores[node] = s + b2[0];
}

// ------------------------------------------------ segment softmax (sorted batch)
__global__ __launch_bounds__(256)
void seg_softmax_kernel(const float* __restrict__ scores, const int* __restrict__ batch,
                        float* __restrict__ wout, int N) {
    const int g = blockIdx.x;
    __shared__ int sS, sE;
    __shared__ float red[256];
    if (threadIdx.x == 0) { sS = lbound(batch, N, g); sE = lbound(batch, N, g + 1); }
    __syncthreads();
    const int s = sS, e = sE;

    float mx = -3.4e38f;
    for (int i = s + threadIdx.x; i < e; i += 256) mx = fmaxf(mx, scores[i]);
    red[threadIdx.x] = mx; __syncthreads();
    for (int st = 128; st; st >>= 1) {
        if (threadIdx.x < st) red[threadIdx.x] = fmaxf(red[threadIdx.x], red[threadIdx.x + st]);
        __syncthreads();
    }
    mx = red[0]; __syncthreads();

    float sm = 0.0f;
    for (int i = s + threadIdx.x; i < e; i += 256) sm += __expf(scores[i] - mx);
    red[threadIdx.x] = sm; __syncthreads();
    for (int st = 128; st; st >>= 1) {
        if (threadIdx.x < st) red[threadIdx.x] += red[threadIdx.x + st];
        __syncthreads();
    }
    const float inv = (red[0] > 0.0f) ? 1.0f / red[0] : 0.0f;
    __syncthreads();
    for (int i = s + threadIdx.x; i < e; i += 256) wout[i] = __expf(scores[i] - mx) * inv;
}

// ------------------------------------------------ gated weighted-sum pooling
__global__ __launch_bounds__(256)
void pool_kernel(const __bf16* __restrict__ h, const float* __restrict__ w,
                 const int* __restrict__ batch, __bf16* __restrict__ pooled,
                 int N, int Kpad) {
    const int g = blockIdx.x;
    __shared__ int sS, sE;
    if (threadIdx.x == 0) { sS = lbound(batch, N, g); sE = lbound(batch, N, g + 1); }
    __syncthreads();
    const int s = sS, e = sE;
    for (int c = threadIdx.x; c < Kpad; c += 256) {
        float acc = 0.0f;
        for (int i = s; i < e; ++i) acc += w[i] * (float)h[(size_t)i * Kpad + c];
        pooled[(size_t)g * Kpad + c] = (__bf16)acc;
    }
}

// ---------------------------------------------------------------- launcher
extern "C" void kernel_launch(void* const* d_in, const int* in_sizes, int n_in,
                              void* d_out, int out_size, void* d_ws, size_t ws_size,
                              hipStream_t stream) {
    (void)in_sizes; (void)n_in; (void)out_size; (void)ws_size;
    constexpr int N = 262144, G = 1024, C = 459, CP = 512, H = 256, O = 128;

    // ---- inputs (setup_inputs dict order, flattened)
    const float* x    = (const float*)d_in[0];
    const int*   bat  = (const int*)  d_in[1];
    const float* encW = (const float*)d_in[2];
    const float* encB = (const float*)d_in[3];
    const float* encG = (const float*)d_in[4];
    const float* encBe= (const float*)d_in[5];
    const float* encM = (const float*)d_in[6];
    const float* encV = (const float*)d_in[7];
    const float* gW1  = (const float*)d_in[8];
    const float* gb1  = (const float*)d_in[9];
    const float* gGa  = (const float*)d_in[10];
    const float* gBe  = (const float*)d_in[11];
    const float* gMe  = (const float*)d_in[12];
    const float* gVa  = (const float*)d_in[13];
    const float* gW2  = (const float*)d_in[14];
    const float* gb2  = (const float*)d_in[15];
    const float* mW[4] = {(const float*)d_in[16], (const float*)d_in[22],
                          (const float*)d_in[28], (const float*)d_in[34]};
    const float* mb[4] = {(const float*)d_in[17], (const float*)d_in[23],
                          (const float*)d_in[29], (const float*)d_in[35]};
    const float* mGa[4]= {(const float*)d_in[18], (const float*)d_in[24],
                          (const float*)d_in[30], (const float*)d_in[36]};
    const float* mBe[4]= {(const float*)d_in[19], (const float*)d_in[25],
                          (const float*)d_in[31], (const float*)d_in[37]};
    const float* mMe[4]= {(const float*)d_in[20], (const float*)d_in[26],
                          (const float*)d_in[32], (const float*)d_in[38]};
    const float* mVa[4]= {(const float*)d_in[21], (const float*)d_in[27],
                          (const float*)d_in[33], (const float*)d_in[39]};

    // ---- workspace carve
    char* ws = (char*)d_ws;
    size_t off = 0;
    auto carve = [&](size_t bytes) -> char* {
        char* p = ws + off;
        off += (bytes + 255) & ~(size_t)255;
        return p;
    };
    __bf16* bufA   = (__bf16*)carve((size_t)N * CP * 2);
    __bf16* bufB   = (__bf16*)carve((size_t)N * CP * 2);
    float*  scores = (float*) carve((size_t)N * 4);
    float*  wAtt   = (float*) carve((size_t)N * 4);
    __bf16* pooled = (__bf16*)carve((size_t)G * CP * 2);
    __bf16* z1     = (__bf16*)carve((size_t)G * H * 2);
    __bf16* z2     = (__bf16*)carve((size_t)G * H * 2);
    __bf16* encBt  = (__bf16*)carve(4ull * CP * CP * 2);
    __bf16* gateBt = (__bf16*)carve((size_t)CP * CP * 2);
    __bf16* m1Bt   = (__bf16*)carve((size_t)H * CP * 2);
    __bf16* m2Bt   = (__bf16*)carve((size_t)H * H * 2);
    __bf16* m3Bt   = (__bf16*)carve((size_t)H * H * 2);
    __bf16* m4Bt   = (__bf16*)carve((size_t)O * H * 2);
    float*  alEnc  = (float*) carve(4ull * CP * 4);
    float*  beEnc  = (float*) carve(4ull * CP * 4);
    float*  alGate = (float*) carve((size_t)CP * 4);
    float*  beGate = (float*) carve((size_t)CP * 4);
    float*  alM[4], *beM[4];
    for (int i = 0; i < 4; ++i) { alM[i] = (float*)carve(CP * 4); beM[i] = (float*)carve(CP * 4); }

    // ---- prep: pack x, transpose+pack weights, fold BN coefficients
    {
        long long tot = (long long)N * CP;
        pack_x_kernel<<<(unsigned)((tot + 255) / 256), 256, 0, stream>>>(x, bufA, tot, C, CP);
    }
    for (int l = 0; l < 4; ++l) {
        pack_wt_kernel<<<(CP * CP + 255) / 256, 256, 0, stream>>>(
            encW + (size_t)l * C * C, encBt + (size_t)l * CP * CP, C, C, CP, CP);
        make_coef_kernel<<<(CP + 255) / 256, 256, 0, stream>>>(
            encB + l * C, encG + l * C, encBe + l * C, encM + l * C, encV + l * C,
            alEnc + l * CP, beEnc + l * CP, C, CP);
    }
    pack_wt_kernel<<<(CP * CP + 255) / 256, 256, 0, stream>>>(gW1, gateBt, C, C, CP, CP);
    make_coef_kernel<<<(CP + 255) / 256, 256, 0, stream>>>(gb1, gGa, gBe, gMe, gVa,
                                                           alGate, beGate, C, CP);
    pack_wt_kernel<<<(H * CP + 255) / 256, 256, 0, stream>>>(mW[0], m1Bt, C, H, CP, H);
    pack_wt_kernel<<<(H * H + 255) / 256, 256, 0, stream>>>(mW[1], m2Bt, H, H, H, H);
    pack_wt_kernel<<<(H * H + 255) / 256, 256, 0, stream>>>(mW[2], m3Bt, H, H, H, H);
    pack_wt_kernel<<<(O * H + 255) / 256, 256, 0, stream>>>(mW[3], m4Bt, H, O, H, O);
    int mc[4] = {H, H, H, O};
    for (int i = 0; i < 4; ++i)
        make_coef_kernel<<<(CP + 255) / 256, 256, 0, stream>>>(
            mb[i], mGa[i], mBe[i], mMe[i], mVa[i], alM[i], beM[i], mc[i], mc[i]);

    // ---- encoder: 4x (GEMM [N,512]x[512,512] + BN + ReLU), ping-pong bf16
    dim3 gridEnc(N / 128, CP / 128);
    __bf16* pin = bufA;
    __bf16* pout = bufB;
    for (int l = 0; l < 4; ++l) {
        gemm_bn_act<<<gridEnc, 256, 0, stream>>>(pin, encBt + (size_t)l * CP * CP,
                                                 alEnc + l * CP, beEnc + l * CP,
                                                 pout, nullptr, N, CP, CP, 1);
        __bf16* t = pin; pin = pout; pout = t;
    }
    __bf16* hFinal = pin;       // after 4 swaps: bufA
    __bf16* gBuf   = pout;      // bufB

    // ---- gate: g = relu(bn(h@W1+b1)); scores = g@W2 + b2
    gemm_bn_act<<<gridEnc, 256, 0, stream>>>(hFinal, gateBt, alGate, beGate,
                                             gBuf, nullptr, N, CP, CP, 1);
    gate_scores_kernel<<<(N * 32) / 256, 256, 0, stream>>>(gBuf, gW2, gb2, scores, N, CP, C);

    // ---- segment softmax + gated pooling
    seg_softmax_kernel<<<G, 256, 0, stream>>>(scores, bat, wAtt, N);
    pool_kernel<<<G, 256, 0, stream>>>(hFinal, wAtt, bat, pooled, N, CP);

    // ---- output MLP on [G=1024, ...]
    dim3 gridM1(G / 128, H / 128);  // Npad=256
    gemm_bn_act<<<gridM1, 256, 0, stream>>>(pooled, m1Bt, alM[0], beM[0],
                                            z1, nullptr, G, CP, H, 1);
    gemm_bn_act<<<gridM1, 256, 0, stream>>>(z1, m2Bt, alM[1], beM[1],
                                            z2, nullptr, G, H, H, 1);
    gemm_bn_act<<<gridM1, 256, 0, stream>>>(z2, m3Bt, alM[2], beM[2],
                                            z1, nullptr, G, H, H, 1);
    dim3 gridM4(G / 128, O / 128);  // Npad=128, fp32 out, no ReLU
    gemm_bn_act<<<gridM4, 256, 0, stream>>>(z1, m4Bt, alM[3], beM[3],
                                            nullptr, (float*)d_out, G, H, O, 0);
}
